// AdditiveAttention_47811575939372
// MI455X (gfx1250) — matrix operand order repaired and checked by reading, not compiled, not run on hardware
//
#include <hip/hip_runtime.h>
#include <hip/hip_bf16.h>
#include <math.h>
#include <stdint.h>

// ---- CDNA5 vector types -----------------------------------------------------
typedef __attribute__((ext_vector_type(16))) _Float16 v16h;
typedef __attribute__((ext_vector_type(8)))  _Float16 v8h;
typedef __attribute__((ext_vector_type(8)))  float    v8f;

// ---- gfx1250 tanh: builtin if present, else raw V_TANH_F32 ------------------
__device__ __forceinline__ float fast_tanh(float x) {
#if __has_builtin(__builtin_amdgcn_tanhf)
  return __builtin_amdgcn_tanhf(x);
#elif __has_builtin(__builtin_amdgcn_tanh_f32)
  return __builtin_amdgcn_tanh_f32(x);
#else
  float y;
  // TRANS op: add a v_nop to satisfy the 1-op TRANS RAW hazard window.
  asm volatile("v_tanh_f32 %0, %1\n\tv_nop" : "=v"(y) : "v"(x));
  return y;
#endif
}

// ---- gfx1250 async global->LDS copy (ASYNCcnt path) -------------------------
// the LDS address comes from the VDST VGPR; generic shared pointers keep the LDS byte
// offset in addr[31:0], so truncation gives the VDST value.
__device__ __forceinline__ uint32_t lds_off32(const void* p) {
  return (uint32_t)(uintptr_t)p;
}
__device__ __forceinline__ void async_copy_b128(uint32_t lds_off, const void* gptr) {
  asm volatile("global_load_async_to_lds_b128 %0, %1, off"
               :: "v"(lds_off), "v"((uint64_t)(uintptr_t)gptr)
               : "memory");
}
__device__ __forceinline__ void wait_asynccnt0() {
  asm volatile("s_wait_asynccnt 0x0" ::: "memory");
}

static constexpr int BB = 16;   // batch
static constexpr int QQ = 128;  // queries
static constexpr int KK = 512;  // keys
static constexpr int HH = 256;  // hidden
static constexpr int VV = 256;  // value dim

// ============================================================================
// Kernel 1: projection GEMM  Y[m,n] = sum_d X[m,d] * W[n,d]   (d,n = 256)
// Workgroup = 128 threads (4 waves), tile 64(M) x 64(N), k-step 32 via
// v_wmma_f32_16x16x32_f16. f32 -> f16 conversion happens while staging to LDS.
// ============================================================================
__global__ __launch_bounds__(128)
void proj_wmma_f16(const float* __restrict__ X, const float* __restrict__ W,
                   float* __restrict__ Y) {
  __shared__ alignas(128) _Float16 As[64][32];   // [m][k]  (k contiguous)
  __shared__ alignas(128) _Float16 Bs[64][32];   // [n][k]  (k contiguous)

  const int tid  = threadIdx.x;
  const int lane = tid & 31;
  const int wave = tid >> 5;
  const int m0   = blockIdx.x * 64;
  const int n0   = blockIdx.y * 64;
  const int mrow = lane & 15;
  const int hsel = lane >> 4;

  v8f acc[4] = {};

  for (int kk = 0; kk < HH; kk += 32) {
#pragma unroll
    for (int i = 0; i < 16; ++i) {
      int idx = tid + 128 * i;
      int r = idx >> 5, c = idx & 31;
      As[r][c] = (_Float16)X[(size_t)(m0 + r) * HH + kk + c];
      Bs[r][c] = (_Float16)W[(size_t)(n0 + r) * HH + kk + c];
    }
    __syncthreads();

    // A fragment: lanes 0-15 K={0..15}, lanes 16-31 K={16..31}
    union { v16h v; v8h h[2]; } a;
    a.h[0] = *(const v8h*)&As[wave * 16 + mrow][hsel * 16];
    a.h[1] = *(const v8h*)&As[wave * 16 + mrow][hsel * 16 + 8];

#pragma unroll
    for (int t = 0; t < 4; ++t) {
      v16h bf = *(const v16h*)&Bs[t * 16 + mrow][hsel * 16];
      acc[t] = __builtin_amdgcn_wmma_f32_16x16x32_f16(
          false, a.v, false, bf, (short)0, acc[t], false, false);
    }
    __syncthreads();
  }

#pragma unroll
  for (int t = 0; t < 4; ++t) {
    int n = n0 + t * 16 + mrow;
#pragma unroll
    for (int r = 0; r < 8; ++r) {
      int m = m0 + wave * 16 + r + hsel * 8;
      Y[(size_t)m * HH + n] = acc[t][r];
    }
  }
}

// ============================================================================
// Kernel 2: fused additive-attention core.
// One workgroup (256 threads, 8 waves) per (batch b, 16-query tile).
//   Phase 1: scores = sum_h Wv[h]*tanh(qp+kp); kp staged via double-buffered
//            global_load_async_to_lds_b128 (ASYNCcnt), tanh on v_tanh_f32.
//   Phase 2: masked softmax over K, weights -> f16 (reuses stage buffer 0).
//   Phase 3: out = attn @ values via WMMA; value tiles staged f32->f16.
// ============================================================================
__global__ __launch_bounds__(256)
void addattn_fused(const float* __restrict__ qp, const float* __restrict__ kp,
                   const float* __restrict__ values,
                   const float* __restrict__ Wv,
                   const int* __restrict__ valid_lens,
                   float* __restrict__ out) {
  __shared__ alignas(128) float sc[16][KK];              // 32 KB (scores)
  __shared__ alignas(128) unsigned char stage0[32768];   // 32 KB
  __shared__ alignas(128) unsigned char stage1[32768];   // 32 KB

  // phase-1 views: two key-chunk buffers [32 keys][256 h] f32
  float (*kbuf0)[HH] = (float (*)[HH])stage0;
  float (*kbuf1)[HH] = (float (*)[HH])stage1;
  // phase-2/3 views (aliases stage0; only valid after phase-1 barrier):
  _Float16 (*attnh)[KK] = (_Float16 (*)[KK])stage0;            // 16 KB
  _Float16 (*vt)[32]    = (_Float16 (*)[32])(stage0 + 16384);  // 16 KB

  const int tid  = threadIdx.x;
  const int lane = tid & 31;
  const int wave = tid >> 5;
  const int b    = blockIdx.x >> 3;
  const int q0   = (blockIdx.x & 7) * 16;

  const float* qpb = qp + ((size_t)b * QQ + q0) * HH;  // [16][256]
  const float* kpb = kp + (size_t)b * KK * HH;         // [512][256]

  // ---- Phase 1: scores, async-staged keys -------------------------------
  float wv[8];
#pragma unroll
  for (int i = 0; i < 8; ++i) wv[i] = Wv[lane + 32 * i];

  // issue chunk 0 into kbuf0 (32 rows x 256 f32 = 2048 x b128, 8 per thread)
#pragma unroll
  for (int i = 0; i < 8; ++i) {
    int idx = tid + 256 * i;                 // 16 bytes each
    int row = idx >> 6;                      // 64 xfers per row
    int col = (idx & 63) << 2;               // float column
    async_copy_b128(lds_off32(&kbuf0[row][col]),
                    kpb + (size_t)row * HH + col);
  }
  wait_asynccnt0();
  __syncthreads();

  for (int kc = 0; kc < KK; kc += 32) {
    float (*cur)[HH] = ((kc >> 5) & 1) ? kbuf1 : kbuf0;
    float (*nxt)[HH] = ((kc >> 5) & 1) ? kbuf0 : kbuf1;

    if (kc + 32 < KK) {   // prefetch next chunk while computing this one
#pragma unroll
      for (int i = 0; i < 8; ++i) {
        int idx = tid + 256 * i;
        int row = idx >> 6;
        int col = (idx & 63) << 2;
        async_copy_b128(lds_off32(&nxt[row][col]),
                        kpb + (size_t)(kc + 32 + row) * HH + col);
      }
    }

    // compute scores for the 32 staged keys; 4 keys per wave
#pragma unroll
    for (int qt = 0; qt < 4; ++qt) {
      float qv[4][8];
#pragma unroll
      for (int j = 0; j < 4; ++j)
#pragma unroll
        for (int i = 0; i < 8; ++i)
          qv[j][i] = qpb[(qt * 4 + j) * HH + lane + 32 * i];

#pragma unroll
      for (int kl = 0; kl < 4; ++kl) {
        int k = wave * 4 + kl;               // local key 0..31
        float krow[8];
#pragma unroll
        for (int i = 0; i < 8; ++i) krow[i] = cur[k][lane + 32 * i];

#pragma unroll
        for (int j = 0; j < 4; ++j) {
          float acc = 0.0f;
#pragma unroll
          for (int i = 0; i < 8; ++i)
            acc = fmaf(wv[i], fast_tanh(qv[j][i] + krow[i]), acc);
#pragma unroll
          for (int off = 16; off; off >>= 1)
            acc += __shfl_xor(acc, off, 32);
          if (lane == 0) sc[qt * 4 + j][kc + k] = acc;
        }
      }
    }

    if (kc + 32 < KK) wait_asynccnt0();
    __syncthreads();   // done reading cur; nxt fully staged for everyone
  }

  // ---- Phase 2: masked softmax over K, store f16 weights ----------------
  const int vl = valid_lens[b];
  for (int q = wave; q < 16; q += 8) {
    float mx = -3.4e38f;
    for (int k = lane; k < KK; k += 32) {
      float s = (k < vl) ? sc[q][k] : -1.0e6f;
      sc[q][k] = s;
      mx = fmaxf(mx, s);
    }
#pragma unroll
    for (int off = 16; off; off >>= 1)
      mx = fmaxf(mx, __shfl_xor(mx, off, 32));

    float sum = 0.0f;
    for (int k = lane; k < KK; k += 32) {
      float e = __expf(sc[q][k] - mx);
      sum += e;
      sc[q][k] = e;
    }
#pragma unroll
    for (int off = 16; off; off >>= 1)
      sum += __shfl_xor(sum, off, 32);
    float inv = 1.0f / sum;

    for (int k = lane; k < KK; k += 32)
      attnh[q][k] = (_Float16)(sc[q][k] * inv);
  }
  __syncthreads();

  // ---- Phase 3: out = attn @ values  (WMMA, K=512 in steps of 32) -------
  const int mrow = lane & 15;
  const int hsel = lane >> 4;
  v8f acc0 = {}, acc1 = {};

  for (int kk = 0; kk < KK; kk += 32) {
    // stage values[kk..kk+31][0..255] transposed into vt[n][k], f32 -> f16
#pragma unroll
    for (int i = 0; i < 32; ++i) {
      int idx = tid + 256 * i;
      int kt = idx >> 8, n = idx & 255;
      vt[n][kt] = (_Float16)values[((size_t)b * KK + kk + kt) * VV + n];
    }
    __syncthreads();

    union { v16h v; v8h h[2]; } a;
    a.h[0] = *(const v8h*)&attnh[mrow][kk + hsel * 16];
    a.h[1] = *(const v8h*)&attnh[mrow][kk + hsel * 16 + 8];

    {
      int n = (wave * 2 + 0) * 16 + mrow;
      v16h bf = *(const v16h*)&vt[n][hsel * 16];
      acc0 = __builtin_amdgcn_wmma_f32_16x16x32_f16(
          false, a.v, false, bf, (short)0, acc0, false, false);
      n = (wave * 2 + 1) * 16 + mrow;
      bf = *(const v16h*)&vt[n][hsel * 16];
      acc1 = __builtin_amdgcn_wmma_f32_16x16x32_f16(
          false, a.v, false, bf, (short)0, acc1, false, false);
    }
    __syncthreads();
  }

#pragma unroll
  for (int r = 0; r < 8; ++r) {
    int m = q0 + r + hsel * 8;
    int n0c = (wave * 2) * 16 + mrow;
    out[((size_t)b * QQ + m) * VV + n0c] = acc0[r];
    out[((size_t)b * QQ + m) * VV + n0c + 16] = acc1[r];
  }
}

// ============================================================================
extern "C" void kernel_launch(void* const* d_in, const int* in_sizes, int n_in,
                              void* d_out, int out_size, void* d_ws, size_t ws_size,
                              hipStream_t stream) {
  (void)in_sizes; (void)n_in; (void)out_size; (void)ws_size;
  const float* queries = (const float*)d_in[0];   // [16,128,256]
  const float* keys    = (const float*)d_in[1];   // [16,512,256]
  const float* values  = (const float*)d_in[2];   // [16,512,256]
  const float* Wq      = (const float*)d_in[3];   // [256,256]
  const float* Wk      = (const float*)d_in[4];   // [256,256]
  const float* Wv      = (const float*)d_in[5];   // [256]
  const int*   vlen    = (const int*)d_in[6];     // [16]
  float* out = (float*)d_out;                     // [16,128,256]

  float* wsf = (float*)d_ws;
  float* qp = wsf;                                // [2048][256] = 2 MB
  float* kp = wsf + (size_t)BB * QQ * HH;         // [8192][256] = 8 MB

  proj_wmma_f16<<<dim3((BB * QQ) / 64, HH / 64), 128, 0, stream>>>(queries, Wq, qp);
  proj_wmma_f16<<<dim3((BB * KK) / 64, HH / 64), 128, 0, stream>>>(keys, Wk, kp);
  addattn_fused<<<dim3(BB * (QQ / 16)), 256, 0, stream>>>(qp, kp, values, Wv, vlen, out);
}
